// TransformerBlock_20057497272503
// MI455X (gfx1250) — matrix operand order repaired and checked by reading, not compiled, not run on hardware
//
#include <hip/hip_runtime.h>

#define BB   4
#define SS   4096
#define EE   512
#define HH   8
#define DD   64
#define NBLK 64          // S / BS
#define RR   3
#define FFN  2048
#define MM   (BB*SS)     // 16384
#define NEGF (-10000.0f)
#define SCALE 0.125f     // D^-0.5

typedef __attribute__((ext_vector_type(16))) __bf16 v16bf;
typedef __attribute__((ext_vector_type(8)))  __bf16 v8bf;
typedef __attribute__((ext_vector_type(4)))  __bf16 v4bf;
typedef __attribute__((ext_vector_type(8)))  float  v8f;

#define WMMA_BF16(A,B,C) \
  __builtin_amdgcn_wmma_f32_16x16x32_bf16(false,(A),false,(B),(short)0,(C),false,false)

__device__ __forceinline__ v4bf cvt4(float4 f) {
  v4bf t;
  t[0] = (__bf16)f.x; t[1] = (__bf16)f.y; t[2] = (__bf16)f.z; t[3] = (__bf16)f.w;
  return t;
}

// A fragment (16x32 bf16): lane half g reads K = kbase+8g..+7 and K = kbase+16+8g..+7
// -> two 16B-contiguous runs -> two ds_load_b128.
__device__ __forceinline__ v16bf frag_ld_A(const __bf16* lds, int row_base,
                                           int stride, int kbase) {
  int lane = threadIdx.x & 31;
  int g    = lane >> 4;
  int row  = row_base + (lane & 15);
  const __bf16* p = lds + row * stride + kbase + 8 * g;
  v8bf lo = *reinterpret_cast<const v8bf*>(p);
  v8bf hi = *reinterpret_cast<const v8bf*>(p + 16);
  return __builtin_shufflevector(lo, hi, 0, 1, 2, 3, 4, 5, 6, 7,
                                 8, 9, 10, 11, 12, 13, 14, 15);
}

// B fragment (32x16 bf16) from N-major storage lds[col][k]: lane half g reads
// K = kbase+8g..+7 and K = kbase+16+8g..+7 -> two ds_load_b128.
__device__ __forceinline__ v16bf frag_ld_B(const __bf16* lds, int col_base,
                                           int stride, int kbase) {
  int lane = threadIdx.x & 31;
  int g    = lane >> 4;
  int col  = col_base + (lane & 15);
  const __bf16* p = lds + col * stride + kbase + 8 * g;
  v8bf lo = *reinterpret_cast<const v8bf*>(p);
  v8bf hi = *reinterpret_cast<const v8bf*>(p + 16);
  return __builtin_shufflevector(lo, hi, 0, 1, 2, 3, 4, 5, 6, 7,
                                 8, 9, 10, 11, 12, 13, 14, 15);
}

// ---------------------------------------------------------------------------
// GEMM: C[M,N] = A[M,K] @ B[K,N] + bias (+ReLU), f32 io, bf16 WMMA.
// 128x64 workgroup tile, K-step 64, 8 waves, double-buffered LDS with
// register prefetch; one barrier per K-step.
// ---------------------------------------------------------------------------
template <bool RELU>
__global__ __launch_bounds__(256) void gemm_kernel(const float* __restrict__ A,
                                                   const float* __restrict__ Bm,
                                                   const float* __restrict__ bias,
                                                   float* __restrict__ C,
                                                   int M, int Nn, int K) {
  __shared__ __align__(16) __bf16 sA[2][128 * 64];  // row-major [row][k]
  __shared__ __align__(16) __bf16 sB[2][64 * 64];   // col-major [col][k]
  int tid  = threadIdx.x;
  int wave = tid >> 5;
  int lane = tid & 31;
  int g    = lane >> 4;
  int lcol = lane & 15;
  int m0   = blockIdx.y * 128;
  int n0   = blockIdx.x * 64;
  v8f acc[4] = {};
  float4 pa[8], pb[4];

  auto load_tiles = [&](int k0) {
#pragma unroll
    for (int i = 0; i < 8; ++i) {
      int idx4 = tid + i * 256;
      int r = idx4 >> 4, c4 = idx4 & 15;
      pa[i] = *reinterpret_cast<const float4*>(A + (size_t)(m0 + r) * K + k0 + c4 * 4);
    }
#pragma unroll
    for (int i = 0; i < 4; ++i) {
      int idx4 = tid + i * 256;
      int r = idx4 >> 4, c0 = (idx4 & 15) * 4;
      pb[i] = *reinterpret_cast<const float4*>(Bm + (size_t)(k0 + r) * Nn + n0 + c0);
    }
  };
  auto store_tiles = [&](int buf) {
#pragma unroll
    for (int i = 0; i < 8; ++i) {
      int idx4 = tid + i * 256;
      int r = idx4 >> 4, c4 = idx4 & 15;
      *reinterpret_cast<v4bf*>(&sA[buf][r * 64 + c4 * 4]) = cvt4(pa[i]);
    }
#pragma unroll
    for (int i = 0; i < 4; ++i) {
      int idx4 = tid + i * 256;
      int r = idx4 >> 4, c0 = (idx4 & 15) * 4;
      sB[buf][(c0 + 0) * 64 + r] = (__bf16)pb[i].x;
      sB[buf][(c0 + 1) * 64 + r] = (__bf16)pb[i].y;
      sB[buf][(c0 + 2) * 64 + r] = (__bf16)pb[i].z;
      sB[buf][(c0 + 3) * 64 + r] = (__bf16)pb[i].w;
    }
  };

  load_tiles(0);
  store_tiles(0);
  __syncthreads();
  int nsteps = K >> 6;
  for (int s = 0; s < nsteps; ++s) {
    int cur = s & 1;
    if (s + 1 < nsteps) load_tiles((s + 1) << 6);   // overlap with WMMAs below
    v16bf a0 = frag_ld_A(sA[cur], wave * 16, 64, 0);
    v16bf a1 = frag_ld_A(sA[cur], wave * 16, 64, 32);
#pragma unroll
    for (int t = 0; t < 4; ++t) {
      v16bf b0 = frag_ld_B(sB[cur], t * 16, 64, 0);
      v16bf b1 = frag_ld_B(sB[cur], t * 16, 64, 32);
      acc[t] = WMMA_BF16(a0, b0, acc[t]);
      acc[t] = WMMA_BF16(a1, b1, acc[t]);
    }
    if (s + 1 < nsteps) store_tiles(cur ^ 1);
    __syncthreads();
  }
#pragma unroll
  for (int t = 0; t < 4; ++t) {
    int gc = n0 + t * 16 + lcol;
    float bi = bias[gc];
#pragma unroll
    for (int r = 0; r < 8; ++r) {
      int grow = m0 + wave * 16 + 8 * g + r;
      float v0 = acc[t][r] + bi;
      if (RELU) v0 = fmaxf(v0, 0.f);
      C[(size_t)grow * Nn + gc] = v0;
    }
  }
}

// key-block schedule, computed arithmetically (pure uniform scalar ops)
__device__ __forceinline__ void bb_sched(int it, int n, int nfix, bool full,
                                         const int* __restrict__ rnd, int h,
                                         int& kb, int& typ, int& boff) {
  typ = 0; boff = 0;
  if (full) {
    kb = it;
  } else if (it >= nfix) {
    kb = rnd[(h * (NBLK - 2) + (n - 1)) * RR + (it - nfix)];
    typ = 2;
  } else if (it == 0) {
    kb = 0;
  } else if (n == 1) {
    kb = (it == 3) ? NBLK - 1 : it;
  } else if (n == NBLK - 2) {
    kb = NBLK - 4 + it;
  } else if (it == 4) {
    kb = NBLK - 1;
  } else {                                  // band blocks n-1, n, n+1
    kb = n - 2 + it;
    typ = 1;
    boff = (it - 1) * 64;
  }
}

// ---------------------------------------------------------------------------
// BigBird attention, one workgroup per (b,h,block). Flash-style online
// softmax over 64-key blocks; O lives in the waves' WMMA C registers.
// K/V for the next block are prefetched into registers during softmax/PV.
// q/k/v layout: [(b*S+s)*512 + h*64 + d] (bshd). ctx same layout.
// ---------------------------------------------------------------------------
__global__ __launch_bounds__(256) void attn_kernel(
    const float* __restrict__ q, const float* __restrict__ k,
    const float* __restrict__ v, const float* __restrict__ band,
    const float* __restrict__ fmask, const float* __restrict__ tmask,
    const float* __restrict__ bmk, const int* __restrict__ rnd,
    float* __restrict__ ctx) {
  int bid = blockIdx.x;
  int n = bid % NBLK;
  int h = (bid / NBLK) % HH;
  int b = bid / (NBLK * HH);

  __shared__ __align__(16) __bf16 sQ[64 * 64];   // [query][d]
  __shared__ __align__(16) __bf16 sK[64 * 64];   // [key][d]   (B for Q@K^T)
  __shared__ __align__(16) __bf16 sVT[64 * 64];  // [d][key]   (B for P@V)
  __shared__ __align__(16) __bf16 sP[64 * 64];   // [query][key]
  __shared__ float sS[64 * 64];
  __shared__ float red[256];
  __shared__ float rowm[64], rowl[64], rowa[64];

  int tid  = threadIdx.x;
  int wave = tid >> 5;
  int lane = tid & 31;
  int g    = lane >> 4;
  int lcol = lane & 15;
  int mi   = wave >> 1;
  int nh   = wave & 1;

  // load Q (scaled) as bf16, vectorized
#pragma unroll
  for (int i = 0; i < 4; ++i) {
    int idx4 = tid + i * 256;
    int qi = idx4 >> 4, c0 = (idx4 & 15) * 4;
    float4 fv = *reinterpret_cast<const float4*>(
        q + ((size_t)(b * SS + n * 64 + qi)) * 512 + h * 64 + c0);
    fv.x *= SCALE; fv.y *= SCALE; fv.z *= SCALE; fv.w *= SCALE;
    *reinterpret_cast<v4bf*>(sQ + qi * 64 + c0) = cvt4(fv);
  }
  if (tid < 64) { rowm[tid] = -1e30f; rowl[tid] = 0.f; }

  v8f accO0 = {}, accO1 = {};

  bool full = (n == 0) || (n == NBLK - 1);
  bool mid  = (n > 1) && (n < NBLK - 2);
  int nfix  = mid ? 5 : 4;
  int nk    = full ? NBLK : (nfix + RR);

  float4 pk[4], pv[4];
  auto prefetch_kv = [&](int kb) {
#pragma unroll
    for (int i = 0; i < 4; ++i) {
      int idx4 = tid + i * 256;
      int kj = idx4 >> 4, c0 = (idx4 & 15) * 4;
      size_t gi = ((size_t)(b * SS + kb * 64 + kj)) * 512 + h * 64 + c0;
      pk[i] = *reinterpret_cast<const float4*>(k + gi);
      pv[i] = *reinterpret_cast<const float4*>(v + gi);
    }
  };

  int kb, typ, boff;
  bb_sched(0, n, nfix, full, rnd, h, kb, typ, boff);
  prefetch_kv(kb);

  for (int it = 0; it < nk; ++it) {
    // commit prefetched K [key][d] / V^T [d][key] into LDS
#pragma unroll
    for (int i = 0; i < 4; ++i) {
      int idx4 = tid + i * 256;
      int kj = idx4 >> 4, c0 = (idx4 & 15) * 4;
      *reinterpret_cast<v4bf*>(sK + kj * 64 + c0) = cvt4(pk[i]);
      sVT[(c0 + 0) * 64 + kj] = (__bf16)pv[i].x;
      sVT[(c0 + 1) * 64 + kj] = (__bf16)pv[i].y;
      sVT[(c0 + 2) * 64 + kj] = (__bf16)pv[i].z;
      sVT[(c0 + 3) * 64 + kj] = (__bf16)pv[i].w;
    }
    __syncthreads();

    // scores S = Q @ K^T  (two 16x16 tiles per wave, Kdim=64 -> 2 steps)
    v16bf aq0 = frag_ld_A(sQ, mi * 16, 64, 0);
    v16bf aq1 = frag_ld_A(sQ, mi * 16, 64, 32);
    v16bf bk00 = frag_ld_B(sK, nh * 32, 64, 0);
    v16bf bk01 = frag_ld_B(sK, nh * 32, 64, 32);
    v16bf bk10 = frag_ld_B(sK, nh * 32 + 16, 64, 0);
    v16bf bk11 = frag_ld_B(sK, nh * 32 + 16, 64, 32);
    v8f c0a = {}, c1a = {};
    c0a = WMMA_BF16(aq0, bk00, c0a); c0a = WMMA_BF16(aq1, bk01, c0a);
    c1a = WMMA_BF16(aq0, bk10, c1a); c1a = WMMA_BF16(aq1, bk11, c1a);

    // write scores with mask fused
    int kj0 = nh * 32 + lcol;
    int kj1 = kj0 + 16;
    float mc0 = 1.f, mc1 = 1.f;
    if (typ == 0) {
      const float* tp = tmask + (size_t)b * SS + (size_t)kb * 64;
      mc0 = tp[kj0]; mc1 = tp[kj1];
    } else if (typ == 2) {
      const float* bp = bmk + ((size_t)b * NBLK + kb) * 64;
      mc0 = bp[kj0]; mc1 = bp[kj1];
    }
#pragma unroll
    for (int r = 0; r < 8; ++r) {
      int row = mi * 16 + 8 * g + r;
      float m0, m1;
      if (typ == 1) {
        size_t bbase = (((size_t)b * (NBLK - 4) + (size_t)(n - 2)) * 64 + row) * 192 + boff;
        m0 = band[bbase + kj0]; m1 = band[bbase + kj1];
      } else if (typ == 2) {
        float qm = bmk[((size_t)b * NBLK + n) * 64 + row];
        m0 = qm * mc0; m1 = qm * mc1;
      } else { m0 = mc0; m1 = mc1; }
      sS[row * 64 + kj0] = c0a[r] + (1.f - m0) * NEGF;
      sS[row * 64 + kj1] = c1a[r] + (1.f - m1) * NEGF;
    }

    // prefetch next key block: HBM latency overlaps softmax + PV below
    if (it + 1 < nk) {
      bb_sched(it + 1, n, nfix, full, rnd, h, kb, typ, boff);
      prefetch_kv(kb);
    }
    __syncthreads();

    // online-softmax row max: 256-thread partial reduce (16 floats each)
    {
      int row = tid >> 2, part = tid & 3;
      const float* sr = sS + row * 64 + part * 16;
      float mx = sr[0];
#pragma unroll
      for (int j = 1; j < 16; ++j) mx = fmaxf(mx, sr[j]);
      red[tid] = mx;
    }
    __syncthreads();
    if (tid < 64) {
      float mx = fmaxf(fmaxf(red[tid * 4], red[tid * 4 + 1]),
                       fmaxf(red[tid * 4 + 2], red[tid * 4 + 3]));
      float mnew = fmaxf(rowm[tid], mx);
      rowa[tid] = __expf(rowm[tid] - mnew);
      rowm[tid] = mnew;
    }
    __syncthreads();

    // P = exp(S - m) (bf16) + row-sum partials in the same pass
    {
      int row = tid >> 2, part = tid & 3;
      int base = row * 64 + part * 16;
      float m = rowm[row];
      float s = 0.f;
#pragma unroll
      for (int j = 0; j < 16; ++j) {
        float p = __expf(sS[base + j] - m);
        sP[base + j] = (__bf16)p;
        s += p;
      }
      red[tid] = s;
    }
    __syncthreads();
    if (tid < 64) {
      rowl[tid] = rowl[tid] * rowa[tid] +
                  red[tid * 4] + red[tid * 4 + 1] + red[tid * 4 + 2] + red[tid * 4 + 3];
    }
    __syncthreads();

    // O = alpha*O + P @ V  (C-operand feedback keeps O in regs)
    v16bf ap0 = frag_ld_A(sP, mi * 16, 64, 0);
    v16bf ap1 = frag_ld_A(sP, mi * 16, 64, 32);
    v16bf bv00 = frag_ld_B(sVT, nh * 32, 64, 0);
    v16bf bv01 = frag_ld_B(sVT, nh * 32, 64, 32);
    v16bf bv10 = frag_ld_B(sVT, nh * 32 + 16, 64, 0);
    v16bf bv11 = frag_ld_B(sVT, nh * 32 + 16, 64, 32);
#pragma unroll
    for (int r = 0; r < 8; ++r) {
      float al = rowa[mi * 16 + 8 * g + r];
      accO0[r] *= al;
      accO1[r] *= al;
    }
    accO0 = WMMA_BF16(ap0, bv00, accO0); accO0 = WMMA_BF16(ap1, bv01, accO0);
    accO1 = WMMA_BF16(ap0, bv10, accO1); accO1 = WMMA_BF16(ap1, bv11, accO1);
    __syncthreads();
  }

  // finalize: divide by l, apply from_mask, write ctx (bshd)
  int c0g = nh * 32 + lcol;
  int c1g = c0g + 16;
#pragma unroll
  for (int r = 0; r < 8; ++r) {
    int row = mi * 16 + 8 * g + r;
    float inv = 1.f / rowl[row];
    float fm  = fmask[(size_t)b * SS + n * 64 + row];
    size_t base = ((size_t)(b * SS + n * 64 + row)) * 512 + h * 64;
    ctx[base + c0g] = accO0[r] * inv * fm;
    ctx[base + c1g] = accO1[r] * inv * fm;
  }
}

// ---------------------------------------------------------------------------
// out = LayerNorm(x + y) * g + beta, one 512-wide row per block (256 thr)
// ---------------------------------------------------------------------------
__global__ __launch_bounds__(256) void add_ln_kernel(const float* __restrict__ x,
                                                     const float* __restrict__ y,
                                                     const float* __restrict__ gw,
                                                     const float* __restrict__ bw,
                                                     float* __restrict__ out) {
  __shared__ float red[256];
  int row = blockIdx.x;
  int tid = threadIdx.x;
  const float* px = x + (size_t)row * EE;
  const float* py = y + (size_t)row * EE;
  float a0 = px[tid] + py[tid];
  float a1 = px[tid + 256] + py[tid + 256];
  red[tid] = a0 + a1;
  __syncthreads();
  for (int s = 128; s > 0; s >>= 1) {
    if (tid < s) red[tid] += red[tid + s];
    __syncthreads();
  }
  float mean = red[0] * (1.f / EE);
  __syncthreads();
  float d0 = a0 - mean, d1 = a1 - mean;
  red[tid] = d0 * d0 + d1 * d1;
  __syncthreads();
  for (int s = 128; s > 0; s >>= 1) {
    if (tid < s) red[tid] += red[tid + s];
    __syncthreads();
  }
  float rstd = rsqrtf(red[0] * (1.f / EE) + 1e-6f);
  out[(size_t)row * EE + tid]       = d0 * rstd * gw[tid] + bw[tid];
  out[(size_t)row * EE + tid + 256] = d1 * rstd * gw[tid + 256] + bw[tid + 256];
}

// ---------------------------------------------------------------------------
extern "C" void kernel_launch(void* const* d_in, const int* in_sizes, int n_in,
                              void* d_out, int out_size, void* d_ws, size_t ws_size,
                              hipStream_t stream) {
  (void)in_sizes; (void)n_in; (void)out_size; (void)ws_size;
  const float* x    = (const float*)d_in[0];
  const float* band = (const float*)d_in[1];
  const float* fm   = (const float*)d_in[2];
  const float* tm   = (const float*)d_in[3];
  const float* bmk  = (const float*)d_in[4];
  const float* Wq   = (const float*)d_in[5];
  const float* bq   = (const float*)d_in[6];
  const float* Wk   = (const float*)d_in[7];
  const float* bk   = (const float*)d_in[8];
  const float* Wv   = (const float*)d_in[9];
  const float* bv   = (const float*)d_in[10];
  const float* Wo   = (const float*)d_in[11];
  const float* bo   = (const float*)d_in[12];
  const float* W1   = (const float*)d_in[13];
  const float* b1   = (const float*)d_in[14];
  const float* W2   = (const float*)d_in[15];
  const float* b2   = (const float*)d_in[16];
  const float* g1   = (const float*)d_in[17];
  const float* be1  = (const float*)d_in[18];
  const float* g2   = (const float*)d_in[19];
  const float* be2  = (const float*)d_in[20];
  const int*   rnd  = (const int*)d_in[21];
  float* out = (float*)d_out;
  float* ws  = (float*)d_ws;

  const size_t MSZ = (size_t)MM * EE;  // 16384*512 floats
  float* qb  = ws;
  float* kb  = ws + MSZ;
  float* vb  = ws + 2 * MSZ;
  float* ctx = ws + 3 * MSZ;
  float* attn = qb;   // reuse after attention consumed q
  float* out1 = kb;   // reuse after attention consumed k
  float* hid  = ctx;  // reuse after Wo gemm consumed ctx
  float* ffn  = vb;   // reuse after attention consumed v

  dim3 blk(256);
  // QKV projections (bshd layout: row = b*S+s, col = h*64+d)
  gemm_kernel<false><<<dim3(EE / 64, MM / 128), blk, 0, stream>>>(x, Wq, bq, qb, MM, EE, EE);
  gemm_kernel<false><<<dim3(EE / 64, MM / 128), blk, 0, stream>>>(x, Wk, bk, kb, MM, EE, EE);
  gemm_kernel<false><<<dim3(EE / 64, MM / 128), blk, 0, stream>>>(x, Wv, bv, vb, MM, EE, EE);
  // BigBird attention
  attn_kernel<<<dim3(BB * HH * NBLK), blk, 0, stream>>>(qb, kb, vb, band, fm, tm, bmk, rnd, ctx);
  // output projection
  gemm_kernel<false><<<dim3(EE / 64, MM / 128), blk, 0, stream>>>(ctx, Wo, bo, attn, MM, EE, EE);
  // residual + LN1
  add_ln_kernel<<<dim3(MM), blk, 0, stream>>>(x, attn, g1, be1, out1);
  // FFN, chunked over rows to bound workspace
  const int CH = 4096;
  for (int c = 0; c < MM / CH; ++c) {
    gemm_kernel<true><<<dim3(FFN / 64, CH / 128), blk, 0, stream>>>(
        out1 + (size_t)c * CH * EE, W1, b1, hid, CH, FFN, EE);
    gemm_kernel<false><<<dim3(EE / 64, CH / 128), blk, 0, stream>>>(
        hid, W2, b2, ffn + (size_t)c * CH * EE, CH, EE, FFN);
  }
  // residual + LN2 -> final output
  add_ln_kernel<<<dim3(MM), blk, 0, stream>>>(out1, ffn, g2, be2, out);
}
